// AnchorStore_6330781795014
// MI455X (gfx1250) — matrix operand order repaired and checked by reading, not compiled, not run on hardware
//
#include <hip/hip_runtime.h>
#include <hip/hip_bf16.h>

// Problem constants (from the reference)
#define D_DIM   50257
#define K_ROWS  4096
#define B_COLS  512
#define NCHUNK  1571              // ceil(D/32)
#define DPAD    (NCHUNK * 32)     // 50272, bf16-padded contraction length
#define LDA     40                // LDS A-tile row stride (bf16 elems), conflict-free for b128 reads

typedef __attribute__((ext_vector_type(16))) __bf16 v16bf;
typedef __attribute__((ext_vector_type(8)))  __bf16 v8bf;
typedef __attribute__((ext_vector_type(8)))  float  v8f;

union BF16Frag { v16bf v; v8bf h[2]; };

// ---------------------------------------------------------------------------
// Kernel 1: logq[b][d] = bf16(log(Q[b][d])), zero-padded out to DPAD.
// ---------------------------------------------------------------------------
__global__ void logq_kernel(const float* __restrict__ q, __bf16* __restrict__ logq) {
    size_t idx = (size_t)blockIdx.x * blockDim.x + threadIdx.x;
    const size_t total = (size_t)B_COLS * DPAD;
    if (idx >= total) return;
    int d = (int)(idx % DPAD);
    int b = (int)(idx / DPAD);
    float v = 0.0f;
    if (d < D_DIM) v = __logf(q[(size_t)b * D_DIM + d]);
    logq[idx] = (__bf16)v;
}

// ---------------------------------------------------------------------------
// Kernel 2: C[k][b] = sum_d A[k][d] * logq[b][d]  (bf16 WMMA, fp32 accum)
//           + fused ent_sum[k] = sum_d A[k][d]*log(A[k][d])  (blockIdx.y==0)
// Block: 256 threads (8 wave32). Tile: 64 rows (k) x 256 cols (b).
// Wave w owns cols [n0 + 32w, +32): 4x2 WMMA tiles of 16x16.
// ---------------------------------------------------------------------------
__global__ __launch_bounds__(256) void kl_gemm_kernel(
    const float*  __restrict__ A,
    const __bf16* __restrict__ logq,
    float*        __restrict__ Cws,
    float*        __restrict__ entws) {

    __shared__ __bf16 Als[64 * LDA];
    __shared__ float  entLds[256];

    const int t    = threadIdx.x;
    const int wave = t >> 5;
    const int lane = t & 31;
    const int k0   = blockIdx.x * 64;
    const int n0   = blockIdx.y * 256 + wave * 32;

    // A-staging mapping: 256 threads cover 64 rows x 32 cols (8 per thread)
    const int ar = t >> 2;            // tile row 0..63
    const int ac = (t & 3) * 8;       // tile col 0,8,16,24

    // Fragment lane mappings (16-bit A 16x32 / B 32x16; contiguous k = 16h + s, same for A and B)
    const int fm  = lane & 15;
    const int akb = (lane < 16) ? 0 : 16;   // A: 16 contiguous K starting at akb
    const int bkb = (lane < 16) ? 0 : 16;   // B: 16 contiguous K starting at bkb

    const v8f vzero = {0.f, 0.f, 0.f, 0.f, 0.f, 0.f, 0.f, 0.f};
    v8f acc[4][2];
#pragma unroll
    for (int mt = 0; mt < 4; ++mt)
#pragma unroll
        for (int nt = 0; nt < 2; ++nt) acc[mt][nt] = vzero;

    float ent_acc = 0.0f;
    const bool doEnt = (blockIdx.y == 0);

    const float* arow = A + (size_t)(k0 + ar) * D_DIM + ac;

    for (int ch = 0; ch < NCHUNK; ++ch) {
        const int db = ch * 32;

        // ---- stage A 64x32 fp32 -> bf16 LDS, fuse ent accumulation ----
        float av[8];
        const float* ap = arow + db;
        if (db + 32 <= D_DIM) {
            float4 f0 = *(const float4*)(ap);
            float4 f1 = *(const float4*)(ap + 4);
            av[0] = f0.x; av[1] = f0.y; av[2] = f0.z; av[3] = f0.w;
            av[4] = f1.x; av[5] = f1.y; av[6] = f1.z; av[7] = f1.w;
            if (doEnt) {
#pragma unroll
                for (int i = 0; i < 8; ++i) ent_acc += av[i] * __logf(av[i]);
            }
            if (ch + 1 < NCHUNK) __builtin_prefetch(ap + 32, 0, 3);
        } else {
#pragma unroll
            for (int i = 0; i < 8; ++i) {
                const int d = db + ac + i;
                const float a = (d < D_DIM) ? ap[i] : 0.0f;
                av[i] = a;
                if (doEnt && d < D_DIM) ent_acc += a * __logf(a);
            }
        }
        __syncthreads();                 // previous chunk's LDS reads done
#pragma unroll
        for (int i = 0; i < 8; ++i) Als[ar * LDA + ac + i] = (__bf16)av[i];
        __syncthreads();

        // ---- B fragments: 16 contiguous bf16 per lane, straight from L2 ----
        BF16Frag bfr[2];
#pragma unroll
        for (int nt = 0; nt < 2; ++nt) {
            const __bf16* bp = logq + (size_t)(n0 + nt * 16 + fm) * DPAD + db + bkb;
            bfr[nt].v = *(const v16bf*)(bp);
        }

        // ---- A fragments from LDS: two b128 reads per lane ----
        BF16Frag afr[4];
#pragma unroll
        for (int mt = 0; mt < 4; ++mt) {
            const __bf16* apl = &Als[(mt * 16 + fm) * LDA + akb];
            afr[mt].h[0] = *(const v8bf*)(apl);
            afr[mt].h[1] = *(const v8bf*)(apl + 8);
        }

        // ---- 8 WMMAs per wave per chunk ----
#pragma unroll
        for (int mt = 0; mt < 4; ++mt)
#pragma unroll
            for (int nt = 0; nt < 2; ++nt)
                acc[mt][nt] = __builtin_amdgcn_wmma_f32_16x16x32_bf16(
                    false, afr[mt].v, false, bfr[nt].v,
                    (short)0, acc[mt][nt], false, false);
    }

    // ---- epilogue: C tiles -> workspace (ISA C layout: lane<16 M=r, else M=r+8)
    const int em = (lane < 16) ? 0 : 8;
    const int en = lane & 15;
#pragma unroll
    for (int mt = 0; mt < 4; ++mt) {
#pragma unroll
        for (int nt = 0; nt < 2; ++nt) {
            const int bg = n0 + nt * 16 + en;
#pragma unroll
            for (int r = 0; r < 8; ++r) {
                const int kg = k0 + mt * 16 + r + em;
                Cws[(size_t)kg * B_COLS + bg] = acc[mt][nt][r];
            }
        }
    }

    // ---- ent reduction: 4 partials per row -> entws[k] ----
    if (doEnt) {
        __syncthreads();
        entLds[t] = ent_acc;
        __syncthreads();
        if (t < 64) {
            float s = entLds[4 * t] + entLds[4 * t + 1] +
                      entLds[4 * t + 2] + entLds[4 * t + 3];
            entws[k0 + t] = s;
        }
    }
}

// ---------------------------------------------------------------------------
// Kernel 3: per-query argmin over anchors, gather label.
// kl ~ ent_sum[k] - C[k][b]  (positive scale 1/D dropped: argmin-invariant)
// ---------------------------------------------------------------------------
__global__ void argmin_kernel(const float* __restrict__ Cws,
                              const float* __restrict__ entws,
                              const int*   __restrict__ label,
                              float*       __restrict__ out) {
    const int b = blockIdx.x * blockDim.x + threadIdx.x;
    if (b >= B_COLS) return;
    float best = __builtin_inff();
    int   bi   = 0;
    for (int k = 0; k < K_ROWS; ++k) {
        const float v = entws[k] - Cws[(size_t)k * B_COLS + b];
        if (v < best) { best = v; bi = k; }
    }
    out[b] = (float)label[bi];
}

// ---------------------------------------------------------------------------
extern "C" void kernel_launch(void* const* d_in, const int* in_sizes, int n_in,
                              void* d_out, int out_size, void* d_ws, size_t ws_size,
                              hipStream_t stream) {
    const float* q     = (const float*)d_in[0];   // (B, D)
    const float* A     = (const float*)d_in[1];   // (K, D)
    const int*   label = (const int*)d_in[2];     // (K,)
    float*       out   = (float*)d_out;           // (B,)

    char* ws = (char*)d_ws;
    const size_t LOGQ_BYTES = (size_t)B_COLS * DPAD * sizeof(__bf16);  // ~51.5 MB
    const size_t C_BYTES    = (size_t)K_ROWS * B_COLS * sizeof(float); // 8 MB
    __bf16* logq  = (__bf16*)ws;
    float*  Cws   = (float*)(ws + LOGQ_BYTES);
    float*  entws = (float*)(ws + LOGQ_BYTES + C_BYTES);

    {   // 1) log(Q) -> bf16 workspace (zero-padded to DPAD)
        const size_t total = (size_t)B_COLS * DPAD;
        const int th = 256;
        const int bl = (int)((total + th - 1) / th);
        logq_kernel<<<bl, th, 0, stream>>>(q, logq);
    }
    {   // 2) bf16 WMMA GEMM + fused entropy
        dim3 grid(K_ROWS / 64, 2);
        kl_gemm_kernel<<<grid, 256, 0, stream>>>(A, logq, Cws, entws);
    }
    {   // 3) argmin + label gather
        argmin_kernel<<<2, 256, 0, stream>>>(Cws, entws, label, out);
    }
}
